// Detect_77524159693073
// MI455X (gfx1250) — matrix operand order repaired and checked by reading, not compiled, not run on hardware
//
#include <hip/hip_runtime.h>
#include <hip/hip_bf16.h>

// ---------------------------------------------------------------------------
// YOLO-style Detect head for MI455X (gfx1250, wave32, WMMA).
// Convs run as implicit GEMM on v_wmma_f32_16x16x32_f16, fed from LDS tiles
// staged with b128 loads; all operands pre-converted to f16.
// Block tile 64(M) x 64(N); each wave computes 16x32 (2 accumulators, shared
// A fragment) -> 4 WMMA issues per K-round of 64.
// ---------------------------------------------------------------------------

typedef __attribute__((ext_vector_type(16))) _Float16 v16h;
typedef __attribute__((ext_vector_type(8)))  _Float16 v8h;
typedef __attribute__((ext_vector_type(8)))  float    v8f;

#define NC      80
#define REG     16
#define NA      8400     // 6400 + 1600 + 400 anchors
#define BATCH   4
#define CAND    1024
#define MAXDET  300
#define CONF    0.001f
#define IOU_T   0.6f

#define KSTEP   64       // K staged per LDS round (all conv K are multiples of 64)
#define LDSPAD  72       // row stride in f16 (144 B, multiple of 16 B)

// ---------------------------------------------------------------------------
// f32 -> f16 conversion (element counts here are all multiples of 4)
// ---------------------------------------------------------------------------
__global__ void cvt_f16(const float* __restrict__ src, _Float16* __restrict__ dst, int n)
{
    const int i = (blockIdx.x * blockDim.x + threadIdx.x) * 4;
    if (i + 3 < n) {
        const float4 v = *(const float4*)(src + i);
        dst[i + 0] = (_Float16)v.x;
        dst[i + 1] = (_Float16)v.y;
        dst[i + 2] = (_Float16)v.z;
        dst[i + 3] = (_Float16)v.w;
    } else {
        for (int j = i; j < n; ++j) dst[j] = (_Float16)src[j];
    }
}

// ---------------------------------------------------------------------------
// Conv (1x1 or 3x3, pad = k/2) as implicit GEMM via WMMA.
//   act=1: out = silu(scale*conv + bias)  -> f16 (outh)
//   act=0: out = conv + bias              -> f32 (outf)
// GEMM: M = Cout, N = H*W per batch, K = Cin*k*k (multiple of 64).
// grid = (ceil(Cout/64), ceil(HW/64), BATCH), block = 256.
// ---------------------------------------------------------------------------
__global__ __launch_bounds__(256)
void conv_wmma(const _Float16* __restrict__ in, const _Float16* __restrict__ wgt,
               const float* __restrict__ scale, const float* __restrict__ bias,
               _Float16* __restrict__ outh, float* __restrict__ outf,
               int Cin, int Cout, int H, int W, int ksize, int act)
{
    __shared__ _Float16 As[64][LDSPAD];   // [m][k] rows of weight tile
    __shared__ _Float16 Bs[64][LDSPAD];   // [n][k] rows of im2col input tile

    const int tid  = threadIdx.x;
    const int lane = tid & 31;
    const int wid  = tid >> 5;
    const int half = lane >> 4;
    const int l16  = lane & 15;

    const int HWp   = H * W;
    const int mblk  = blockIdx.x * 64;
    const int nblk  = blockIdx.y * 64;
    const int bz    = blockIdx.z;
    const int pad   = ksize >> 1;
    const int K     = Cin * ksize * ksize;

    // --- staging roles -----------------------------------------------------
    // A: thread t -> row t/4 (0..63), k-chunk (t%4)*16  (two b128 loads)
    const int arow   = tid >> 2;
    const int akc    = (tid & 3) * 16;
    const int am     = mblk + arow;
    const bool amok  = (am < Cout);
    const _Float16* wrow = wgt + (size_t)(amok ? am : 0) * K;

    // B: thread t -> n row t/4 (0..63), k-chunk (t%4)*16 (two b128 stores)
    const int brow   = tid >> 2;
    const int bkc    = (tid & 3) * 16;
    const int bn     = nblk + brow;
    const bool bnok  = (bn < HWp);
    const int bnc    = bnok ? bn : 0;
    const int boy    = bnc / W;
    const int box_   = bnc - boy * W;
    const _Float16* inb = in + (size_t)bz * Cin * HWp;

    // --- compute roles -----------------------------------------------------
    const int mtile = (wid & 3) * 16;     // wave's M offset within block tile
    const int ntile = (wid >> 2) * 32;    // wave's N offset within block tile

    v8f acc0 = {};
    v8f acc1 = {};

    for (int k0 = 0; k0 < K; k0 += KSTEP) {
        // ---------------- stage A (weights): 64 x 64 f16 ------------------
        if (k0 + KSTEP < K) __builtin_prefetch(&wrow[k0 + KSTEP + akc], 0, 1);
        {
            v8h w0, w1;
            if (amok) {
                w0 = *(const v8h*)&wrow[k0 + akc];
                w1 = *(const v8h*)&wrow[k0 + akc + 8];
            } else {
                w0 = (v8h)(_Float16)0.0f;
                w1 = (v8h)(_Float16)0.0f;
            }
            *(v8h*)&As[arow][akc]     = w0;
            *(v8h*)&As[arow][akc + 8] = w1;
        }
        // ---------------- stage B (im2col input): 64 x 64 f16 -------------
        {
            v8h bv0, bv1;
            #pragma unroll
            for (int j = 0; j < 16; ++j) {
                const int kk = k0 + bkc + j;
                _Float16 bv = (_Float16)0.0f;
                if (bnok) {
                    int ci, ky, kx;
                    if (ksize == 3) {
                        ci = kk / 9;
                        const int r = kk - ci * 9;
                        ky = r / 3;
                        kx = r - ky * 3;
                    } else {
                        ci = kk; ky = 0; kx = 0;
                    }
                    const int iy = boy + ky - pad;
                    const int ix = box_ + kx - pad;
                    if (iy >= 0 && iy < H && ix >= 0 && ix < W)
                        bv = inb[((size_t)ci * H + iy) * W + ix];
                }
                if (j < 8) bv0[j] = bv; else bv1[j - 8] = bv;
            }
            *(v8h*)&Bs[brow][bkc]     = bv0;
            *(v8h*)&Bs[brow][bkc + 8] = bv1;
        }
        __syncthreads();

        // ---------------- 4 WMMAs fed by ds_load_b128 ---------------------
        const _Float16* ar  = &As[mtile + l16][0];
        const _Float16* br0 = &Bs[ntile + l16][0];
        const _Float16* br1 = &Bs[ntile + 16 + l16][0];
        #pragma unroll
        for (int ks = 0; ks < KSTEP; ks += 32) {
            // A fragment: lanes h hold k = ks + h*16 + e (contiguous 16)
            const v8h a0 = *(const v8h*)&ar[ks + half * 16];
            const v8h a1 = *(const v8h*)&ar[ks + half * 16 + 8];
            // B fragments: lanes h hold k = ks + h*16 + e (contiguous 16)
            const v8h p0 = *(const v8h*)&br0[ks + half * 16];
            const v8h p1 = *(const v8h*)&br0[ks + half * 16 + 8];
            const v8h q0 = *(const v8h*)&br1[ks + half * 16];
            const v8h q1 = *(const v8h*)&br1[ks + half * 16 + 8];
            v16h a, b0, b1;
            #pragma unroll
            for (int e = 0; e < 8; ++e) {
                a[e]  = a0[e]; a[e + 8]  = a1[e];
                b0[e] = p0[e]; b0[e + 8] = p1[e];
                b1[e] = q0[e]; b1[e + 8] = q1[e];
            }
            acc0 = __builtin_amdgcn_wmma_f32_16x16x32_f16(
                       false, a, false, b0, (short)0, acc0, false, false);
            acc1 = __builtin_amdgcn_wmma_f32_16x16x32_f16(
                       false, a, false, b1, (short)0, acc1, false, false);
        }
        __syncthreads();
    }

    // ---- epilogue: C/D layout (VGPR v, lanes 0-15 -> M=v, 16-31 -> M=v+8)
    #pragma unroll
    for (int v = 0; v < 8; ++v) {
        const int m  = mblk + mtile + v + half * 8;
        const int n0 = nblk + ntile + l16;
        const int n1 = n0 + 16;
        if (m < Cout) {
            const float sc = act ? scale[m] : 1.0f;
            const float bi = bias[m];
            const size_t rowoff = ((size_t)bz * Cout + m) * HWp;
            if (n0 < HWp) {
                float y = acc0[v];
                if (act) {
                    y = y * sc + bi;
                    y = y / (1.0f + __expf(-y));      // SiLU
                    outh[rowoff + n0] = (_Float16)y;
                } else {
                    outf[rowoff + n0] = y + bi;
                }
            }
            if (n1 < HWp) {
                float y = acc1[v];
                if (act) {
                    y = y * sc + bi;
                    y = y / (1.0f + __expf(-y));      // SiLU
                    outh[rowoff + n1] = (_Float16)y;
                } else {
                    outf[rowoff + n1] = y + bi;
                }
            }
        }
    }
}

// ---------------------------------------------------------------------------
// DFL decode + class sigmoid.  f2: per-scale [B][64][HW], f3: [B][80][HW].
// ---------------------------------------------------------------------------
__global__ void decode_kernel(const float* __restrict__ f2,
                              const float* __restrict__ f3,
                              float* __restrict__ boxes,
                              float* __restrict__ smaxA,
                              float* __restrict__ lblA,
                              float* __restrict__ svalA)
{
    const int idx = blockIdx.x * blockDim.x + threadIdx.x;
    if (idx >= BATCH * NA) return;
    const int b = idx / NA;
    const int a = idx - b * NA;

    int s, p, Wd, hw; float stride;
    if (a < 6400)      { s = 0; p = a;        Wd = 80; hw = 6400; stride = 8.0f;  }
    else if (a < 8000) { s = 1; p = a - 6400; Wd = 40; hw = 1600; stride = 16.0f; }
    else               { s = 2; p = a - 8000; Wd = 20; hw = 400;  stride = 32.0f; }

    const size_t off2[3] = {0, (size_t)BATCH*64*6400, (size_t)BATCH*64*(6400+1600)};
    const size_t off3[3] = {0, (size_t)BATCH*NC*6400, (size_t)BATCH*NC*(6400+1600)};
    const float* F2 = f2 + off2[s] + ((size_t)b * 64) * hw + p;
    const float* F3 = f3 + off3[s] + ((size_t)b * NC) * hw + p;

    float d[4];
    #pragma unroll
    for (int side = 0; side < 4; ++side) {
        float lg[REG], mx = -1e30f;
        #pragma unroll
        for (int r = 0; r < REG; ++r) {
            lg[r] = F2[(size_t)(side * REG + r) * hw];
            mx = fmaxf(mx, lg[r]);
        }
        float den = 0.0f, num = 0.0f;
        #pragma unroll
        for (int r = 0; r < REG; ++r) {
            const float e = __expf(lg[r] - mx);
            den += e;
            num += e * (float)r;
        }
        d[side] = num / den;
    }

    const float ax = (float)(p % Wd) + 0.5f;
    const float ay = (float)(p / Wd) + 0.5f;
    float* bo = boxes + (size_t)idx * 4;
    bo[0] = (ax - d[0]) * stride;
    bo[1] = (ay - d[1]) * stride;
    bo[2] = (ax + d[2]) * stride;
    bo[3] = (ay + d[3]) * stride;

    float best = -1e30f; int bi = 0;
    for (int c = 0; c < NC; ++c) {
        const float v = F3[(size_t)c * hw];
        if (v > best) { best = v; bi = c; }
    }
    const float sm = 1.0f / (1.0f + __expf(-best));
    smaxA[idx] = sm;
    lblA[idx]  = (float)bi;
    svalA[idx] = (sm > CONF) ? sm : -1.0f;
}

// ---------------------------------------------------------------------------
// Exact stable top-1024 by rank counting (matches lax.top_k stable ordering).
// ---------------------------------------------------------------------------
__global__ void topk_rank(const float* __restrict__ svalA,
                          const float* __restrict__ boxes,
                          const float* __restrict__ smaxA,
                          const float* __restrict__ lblA,
                          float* __restrict__ cbx, float* __restrict__ csc,
                          float* __restrict__ clb, int* __restrict__ cvd)
{
    const int idx = blockIdx.x * blockDim.x + threadIdx.x;
    if (idx >= BATCH * NA) return;
    const int b = idx / NA;
    const int i = idx - b * NA;
    const float* sv = svalA + (size_t)b * NA;
    const float v = sv[i];
    int rank = 0;
    for (int j = 0; j < NA; ++j) {
        const float vj = sv[j];
        rank += (vj > v) || ((vj == v) && (j < i));
    }
    if (rank < CAND) {
        const int dst = b * CAND + rank;
        const float* bo = boxes + (size_t)idx * 4;
        cbx[(size_t)dst * 4 + 0] = bo[0];
        cbx[(size_t)dst * 4 + 1] = bo[1];
        cbx[(size_t)dst * 4 + 2] = bo[2];
        cbx[(size_t)dst * 4 + 3] = bo[3];
        csc[dst] = smaxA[idx];
        clb[dst] = lblA[idx];
        cvd[dst] = (v > CONF) ? 1 : 0;
    }
}

// ---------------------------------------------------------------------------
// Greedy NMS over 1024 sorted candidates; one 1024-thread block per image.
// ---------------------------------------------------------------------------
__global__ __launch_bounds__(CAND)
void nms_kernel(const float* __restrict__ cbx, const float* __restrict__ csc,
                const float* __restrict__ clb, const int* __restrict__ cvd,
                float* __restrict__ out)
{
    __shared__ float sx1[CAND], sy1[CAND], sx2[CAND], sy2[CAND], sar[CAND];
    __shared__ int   ssup[CAND], skept[CAND], srank[CAND];

    const int t = threadIdx.x;
    const int b = blockIdx.x;
    const int g = b * CAND + t;

    const float x1 = cbx[(size_t)g * 4 + 0];
    const float y1 = cbx[(size_t)g * 4 + 1];
    const float x2 = cbx[(size_t)g * 4 + 2];
    const float y2 = cbx[(size_t)g * 4 + 3];
    const int valid = cvd[g];
    sx1[t] = x1; sy1[t] = y1; sx2[t] = x2; sy2[t] = y2;
    sar[t] = (x2 - x1) * (y2 - y1);
    ssup[t] = valid ? 0 : 1;
    __syncthreads();

    for (int i = 0; i < CAND; ++i) {
        if (ssup[i] == 0 && t > i) {
            const float iw = fmaxf(fminf(x2, sx2[i]) - fmaxf(x1, sx1[i]), 0.0f);
            const float ih = fmaxf(fminf(y2, sy2[i]) - fmaxf(y1, sy1[i]), 0.0f);
            const float inter = iw * ih;
            const float iou = inter / (sar[t] + sar[i] - inter);
            if (iou > IOU_T) ssup[t] = 1;
        }
        __syncthreads();
    }

    skept[t] = (valid && ssup[t] == 0) ? 1 : 0;
    __syncthreads();
    if (t == 0) {
        int r = 0;
        for (int i = 0; i < CAND; ++i) { srank[i] = skept[i] ? r : -1; r += skept[i]; }
    }
    __syncthreads();

    const int r = srank[t];
    if (skept[t] && r < MAXDET) {
        float* row = out + ((size_t)b * MAXDET + r) * 6;
        row[0] = x1; row[1] = y1; row[2] = x2; row[3] = y2;
        row[4] = csc[g]; row[5] = clb[g];
    }
}

__global__ void zero_kernel(float* __restrict__ p, int n)
{
    const int i = blockIdx.x * blockDim.x + threadIdx.x;
    if (i < n) p[i] = 0.0f;
}

// ---------------------------------------------------------------------------
// Host-side orchestration.
// d_in: [0..2] x0,x1,x2 ; [3+8s+k] cv2_params[s] ; [27+8s+k] cv3_params[s]
//       (each branch: w1,s1,b1,w2,s2,b2,wf,bf)
// d_out: (4,300,6) float32 flat.
// ---------------------------------------------------------------------------
extern "C" void kernel_launch(void* const* d_in, const int* in_sizes, int n_in,
                              void* d_out, int out_size, void* d_ws, size_t ws_size,
                              hipStream_t stream)
{
    (void)in_sizes; (void)n_in; (void)ws_size;

    static const int CHs[3] = {256, 512, 512};
    static const int Hs[3]  = {80, 40, 20};

    // ---- f32 workspace region ----
    float* Wf = (float*)d_ws;
    size_t o = 0;
    float* f2    = Wf + o; o += (size_t)BATCH * 64 * NA;
    float* f3    = Wf + o; o += (size_t)BATCH * NC * NA;
    float* boxes = Wf + o; o += (size_t)BATCH * NA * 4;
    float* smaxA = Wf + o; o += (size_t)BATCH * NA;
    float* lblA  = Wf + o; o += (size_t)BATCH * NA;
    float* svalA = Wf + o; o += (size_t)BATCH * NA;
    float* cbx   = Wf + o; o += (size_t)BATCH * CAND * 4;
    float* csc   = Wf + o; o += (size_t)BATCH * CAND;
    float* clb   = Wf + o; o += (size_t)BATCH * CAND;
    int*   cvd   = (int*)(Wf + o); o += (size_t)BATCH * CAND;

    // ---- f16 workspace region (after f32, 16B aligned by construction) ----
    _Float16* Hf = (_Float16*)(Wf + ((o + 3) & ~(size_t)3));
    size_t oh = 0;
    auto allocH = [&](size_t n) { _Float16* p = Hf + oh; oh += (n + 7) & ~(size_t)7; return p; };

    _Float16* xh[3];
    xh[0] = allocH((size_t)BATCH * 256 * 6400);
    xh[1] = allocH((size_t)BATCH * 512 * 1600);
    xh[2] = allocH((size_t)BATCH * 512 * 400);
    _Float16* mid_a = allocH((size_t)BATCH * 256 * 6400);
    _Float16* mid_b = allocH((size_t)BATCH * 256 * 6400);

    const size_t off2[3] = {0, (size_t)BATCH*64*6400, (size_t)BATCH*64*(6400+1600)};
    const size_t off3[3] = {0, (size_t)BATCH*NC*6400, (size_t)BATCH*NC*(6400+1600)};

    float* outf = (float*)d_out;
    zero_kernel<<<(out_size + 255) / 256, 256, 0, stream>>>(outf, out_size);

    // convert input activations to f16
    for (int s = 0; s < 3; ++s) {
        const int n = BATCH * CHs[s] * Hs[s] * Hs[s];
        cvt_f16<<<(n / 4 + 255) / 256, 256, 0, stream>>>((const float*)d_in[s], xh[s], n);
    }

    // convert all weight tensors to f16 (scales/biases stay f32)
    _Float16* whalf[3][2][3];   // [scale][tower][layer]
    for (int s = 0; s < 3; ++s) {
        for (int tw = 0; tw < 2; ++tw) {
            const int base = (tw == 0 ? 3 : 27) + s * 8;
            const int cmid = (tw == 0) ? 64 : 256;
            const int cout = (tw == 0) ? 64 : NC;
            const size_t n1 = (size_t)cmid * CHs[s] * 9;
            const size_t n2 = (size_t)cmid * cmid * 9;
            const size_t n3 = (size_t)cout * cmid;
            whalf[s][tw][0] = allocH(n1);
            whalf[s][tw][1] = allocH(n2);
            whalf[s][tw][2] = allocH(n3);
            cvt_f16<<<((int)n1 / 4 + 255) / 256, 256, 0, stream>>>((const float*)d_in[base + 0], whalf[s][tw][0], (int)n1);
            cvt_f16<<<((int)n2 / 4 + 255) / 256, 256, 0, stream>>>((const float*)d_in[base + 3], whalf[s][tw][1], (int)n2);
            cvt_f16<<<((int)n3 / 4 + 255) / 256, 256, 0, stream>>>((const float*)d_in[base + 6], whalf[s][tw][2], (int)n3);
        }
    }

    // conv towers
    for (int s = 0; s < 3; ++s) {
        const int Cin = CHs[s];
        const int H = Hs[s], W = Hs[s];
        const int hw = H * W;
        const dim3 blk(256);
        const int gy = (hw + 63) / 64;

        for (int tw = 0; tw < 2; ++tw) {
            const int base = (tw == 0 ? 3 : 27) + s * 8;
            const int cmid = (tw == 0) ? 64 : 256;
            const int cout = (tw == 0) ? 64 : NC;
            const float* s1 = (const float*)d_in[base + 1];
            const float* b1 = (const float*)d_in[base + 2];
            const float* s2 = (const float*)d_in[base + 4];
            const float* b2 = (const float*)d_in[base + 5];
            const float* bf = (const float*)d_in[base + 7];
            float* fout = (tw == 0) ? (f2 + off2[s]) : (f3 + off3[s]);

            conv_wmma<<<dim3((cmid + 63) / 64, gy, BATCH), blk, 0, stream>>>(
                xh[s], whalf[s][tw][0], s1, b1, mid_a, nullptr,
                Cin, cmid, H, W, 3, 1);
            conv_wmma<<<dim3((cmid + 63) / 64, gy, BATCH), blk, 0, stream>>>(
                mid_a, whalf[s][tw][1], s2, b2, mid_b, nullptr,
                cmid, cmid, H, W, 3, 1);
            conv_wmma<<<dim3((cout + 63) / 64, gy, BATCH), blk, 0, stream>>>(
                mid_b, whalf[s][tw][2], nullptr, bf, nullptr, fout,
                cmid, cout, H, W, 1, 0);
        }
    }

    const int tot = BATCH * NA;
    decode_kernel<<<(tot + 255) / 256, 256, 0, stream>>>(
        f2, f3, boxes, smaxA, lblA, svalA);
    topk_rank<<<(tot + 255) / 256, 256, 0, stream>>>(
        svalA, boxes, smaxA, lblA, cbx, csc, clb, cvd);
    nms_kernel<<<BATCH, CAND, 0, stream>>>(cbx, csc, clb, cvd, outf);
}
